// NeighborhoodAttentionModule_6923487282208
// MI455X (gfx1250) — matrix operand order repaired and checked by reading, not compiled, not run on hardware
//
#include <hip/hip_runtime.h>

// ---------------- CDNA5 WMMA types ----------------
typedef __attribute__((ext_vector_type(16))) __bf16 v16bf;
typedef __attribute__((ext_vector_type(8)))  float  v8f;

// Problem constants (fixed by the reference)
constexpr int DIM = 256;   // D
constexpr int KNB = 16;    // K neighbors
constexpr int NH  = 2;     // heads
constexpr int HA  = 128;   // H*A

// ---------------- bf16 helpers (bit-level, no scalar __bf16 math) ----------------
__device__ __forceinline__ unsigned short f2bf(float f) {
  unsigned u = __float_as_uint(f);
  return (unsigned short)((u + 0x7FFFu + ((u >> 16) & 1u)) >> 16);  // RNE
}
__device__ __forceinline__ float bf2f(unsigned short h) {
  return __uint_as_float(((unsigned)h) << 16);
}
__device__ __forceinline__ float dot8bf(const float* c, uint4 u) {
  float s;
  s  = c[0] * bf2f((unsigned short)(u.x & 0xffffu));
  s += c[1] * bf2f((unsigned short)(u.x >> 16));
  s += c[2] * bf2f((unsigned short)(u.y & 0xffffu));
  s += c[3] * bf2f((unsigned short)(u.y >> 16));
  s += c[4] * bf2f((unsigned short)(u.z & 0xffffu));
  s += c[5] * bf2f((unsigned short)(u.z >> 16));
  s += c[6] * bf2f((unsigned short)(u.w & 0xffffu));
  s += c[7] * bf2f((unsigned short)(u.w >> 16));
  return s;
}

// ---------------- Prologue: convert weights f32 -> bf16 into workspace ----------------
// layout in d_ws (ushort elems): [Wq 32768][Wk 32768][Wo 65536]  (total 256 KB)
__global__ void convert_weights_kernel(const float* __restrict__ wq,
                                       const float* __restrict__ wk,
                                       const float* __restrict__ wo,
                                       unsigned short* __restrict__ dst) {
  int i = blockIdx.x * blockDim.x + threadIdx.x;
  if (i < 32768)        dst[i] = f2bf(wq[i]);
  else if (i < 65536)   dst[i] = f2bf(wk[i - 32768]);
  else if (i < 131072)  dst[i] = f2bf(wo[i - 65536]);
}

// ---------------- Main kernel: one workgroup (8 wave32) per center node ----------------
__global__ __launch_bounds__(256)
void neigh_attn_kernel(const float* __restrict__ center,
                       const float* __restrict__ all_emb,
                       const int*   __restrict__ nidx,
                       const float* __restrict__ nwt,
                       const unsigned short* __restrict__ wqb,
                       const unsigned short* __restrict__ wkb,
                       const unsigned short* __restrict__ wob,
                       float* __restrict__ out) {
  __shared__ float nbf[KNB][DIM];                           // neighbors f32 (for wv)
  __shared__ __align__(16) unsigned short nbb[KNB][DIM];    // neighbors bf16 (WMMA A)
  __shared__ float kp[KNB][HA];                             // key projections
  __shared__ float cen[DIM];
  __shared__ float qv[HA];
  __shared__ float wts[KNB];
  __shared__ float sc[NH][KNB];
  __shared__ float at[NH][KNB];
  __shared__ float ctx[DIM];

  const int b = blockIdx.x;
  const int t = threadIdx.x;

  // ---- Phase 1: center + weight prefetch + neighbor gather ----
  cen[t] = center[(size_t)b * DIM + t];
  if (t < KNB) wts[t] = nwt[b * KNB + t];
  // warm the bf16 weights (L2-resident) into near caches while gather is in flight
  __builtin_prefetch(wkb + t * 128, 0, 0);   // 256 thr * 256B = all of Wk_bf16
  __builtin_prefetch(wob + t * 256, 0, 0);
  __builtin_prefetch(wob + t * 256 + 128, 0, 0);
  {
    const int r = t >> 4;                 // neighbor row 0..15
    const int c = t & 15;                 // 16-float chunk within row
    const size_t row = (size_t)nidx[b * KNB + r] * DIM;
    const float4* src = (const float4*)(all_emb + row + c * 16);
    #pragma unroll
    for (int i = 0; i < 4; ++i) {
      float4 v = src[i];
      const int d = c * 16 + i * 4;
      nbf[r][d + 0] = v.x; nbf[r][d + 1] = v.y;
      nbf[r][d + 2] = v.z; nbf[r][d + 3] = v.w;
      nbb[r][d + 0] = f2bf(v.x); nbb[r][d + 1] = f2bf(v.y);
      nbb[r][d + 2] = f2bf(v.z); nbb[r][d + 3] = f2bf(v.w);
    }
  }
  __syncthreads();

  // ---- Phase 2a: Q = center @ Wq^T  (threads 0..127, VALU; tiny: 32K MAC) ----
  if (t < HA) {
    const uint4* wr = (const uint4*)(wqb + (size_t)t * DIM);
    float q = 0.f;
    #pragma unroll 4
    for (int j = 0; j < DIM / 8; ++j) q += dot8bf(&cen[j * 8], wr[j]);
    qv[t] = q;
  }

  // ---- Phase 2b: Kp = nb @ Wk^T  via v_wmma_f32_16x16x32_bf16 ----
  // M=16 (neighbors), N=128 (h*a), K=256 (dim). Wave w owns output cols [16w,16w+16).
  {
    const int lane = t & 31;
    const int wave = t >> 5;
    const int half = lane >> 4;      // lane group: selects 16-wide K block (contiguous k order used here)
    const int ln   = lane & 15;      // A: row m ; B/D: col n
    const int n0   = wave * 16;
    union ABfrag { uint4 u[2]; v16bf v; };
    v8f acc = {};
    #pragma unroll
    for (int s = 0; s < 8; ++s) {
      const int d0 = s * 32;
      ABfrag a, bb;
      // A (16x32 bf16): lane-half holds K = d0+16*half+e (e<16)
      a.u[0] = *(const uint4*)&nbb[ln][d0 + 16 * half];
      a.u[1] = *(const uint4*)&nbb[ln][d0 + 16 * half + 8];
      // B (32x16 bf16): B[k][n] = Wk[n0+n][k]; lanes 0-15 K=d0..d0+15, lanes 16-31 K=d0+16..d0+31
      const unsigned short* wrow = wkb + (size_t)(n0 + ln) * DIM + d0 + 16 * half;
      bb.u[0] = *(const uint4*)(wrow);
      bb.u[1] = *(const uint4*)(wrow + 8);
      acc = __builtin_amdgcn_wmma_f32_16x16x32_bf16(false, a.v, false, bb.v,
                                                    (short)0, acc, false, false);
    }
    // D (16x16 f32): VGPR v -> row v+8*half, lane&15 -> col
    #pragma unroll
    for (int v2 = 0; v2 < 8; ++v2) kp[v2 + 8 * half][n0 + ln] = acc[v2];
  }
  __syncthreads();

  // ---- Phase 3: scores + clipped log-weight bias (threads 0..31) ----
  if (t < NH * KNB) {
    const int h = t >> 4, k = t & 15;
    float wsum = 0.f;
    #pragma unroll
    for (int kk = 0; kk < KNB; ++kk) wsum += wts[kk];
    wsum = fmaxf(wsum, 1e-9f);                         // clip(sum, 1e-9)
    float s = 0.f;
    #pragma unroll
    for (int a = 0; a < 64; ++a) s += qv[h * 64 + a] * kp[k][h * 64 + a];
    s *= 0.125f;                                       // 1/sqrt(A), A=64
    const float bias = fmaxf(__logf(wts[k]) - __logf(wsum), -10.0f);
    sc[h][k] = s + bias;
  }
  __syncthreads();

  // ---- Phase 4: softmax over K per head (threads 0..1; K=16, trivial) ----
  if (t < NH) {
    float m = -3.4e38f;
    for (int k = 0; k < KNB; ++k) m = fmaxf(m, sc[t][k]);
    float sum = 0.f;
    for (int k = 0; k < KNB; ++k) { float e = __expf(sc[t][k] - m); at[t][k] = e; sum += e; }
    const float inv = 1.0f / sum;
    for (int k = 0; k < KNB; ++k) at[t][k] *= inv;
  }
  __syncthreads();

  // ---- Phase 5: wv per head + head-mean context (thread t -> dim t) ----
  {
    float wv0 = 0.f, wv1 = 0.f;
    #pragma unroll
    for (int k = 0; k < KNB; ++k) {
      const float nv = nbf[k][t];
      wv0 += at[0][k] * nv;
      wv1 += at[1][k] * nv;
    }
    ctx[t] = 0.5f * (wv0 + wv1);
  }
  __syncthreads();

  // ---- Phase 6: out = center + ctx @ Wo^T ----
  {
    const uint4* wr = (const uint4*)(wob + (size_t)t * DIM);
    float o = 0.f;
    #pragma unroll 4
    for (int j = 0; j < DIM / 8; ++j) o += dot8bf(&ctx[j * 8], wr[j]);
    out[(size_t)b * DIM + t] = cen[t] + o;
  }
}

extern "C" void kernel_launch(void* const* d_in, const int* in_sizes, int n_in,
                              void* d_out, int out_size, void* d_ws, size_t ws_size,
                              hipStream_t stream) {
  const float* center  = (const float*)d_in[0];
  const float* all_emb = (const float*)d_in[1];
  const int*   nidx    = (const int*)d_in[2];
  const float* nwt     = (const float*)d_in[3];
  const float* Wq      = (const float*)d_in[4];
  const float* Wk      = (const float*)d_in[5];
  const float* Wo      = (const float*)d_in[6];
  float* out = (float*)d_out;

  unsigned short* wsb = (unsigned short*)d_ws;
  unsigned short* wqb = wsb;
  unsigned short* wkb = wsb + 32768;
  unsigned short* wob = wsb + 65536;

  const int B = in_sizes[0] / DIM;

  // 131072 weight elements -> 512 blocks of 256
  hipLaunchKernelGGL(convert_weights_kernel, dim3(512), dim3(256), 0, stream,
                     Wq, Wk, Wo, wsb);
  hipLaunchKernelGGL(neigh_attn_kernel, dim3(B), dim3(256), 0, stream,
                     center, all_emb, nidx, nwt, wqb, wkb, wob, out);
}
